// MiniGAT_86388972191824
// MI455X (gfx1250) — compile-verified, matrix operand order checked by reading, not run on hardware
//
#include <hip/hip_runtime.h>
#include <hip/hip_bf16.h>
#include <math.h>

// ---------------- problem constants (match reference) ----------------
#define NNODES  50000
#define NEDGES  800000
#define ETOT    (NEDGES + NNODES)   // self loops appended
#define BGRAPHS 64
#define INDIM   773
#define HIDF    256
#define OUTF    128
#define HEADS   4
#define DIM1    64
#define DIM2    32

typedef __attribute__((ext_vector_type(16))) _Float16 v16h;
typedef __attribute__((ext_vector_type(8)))  _Float16 v8h;
typedef __attribute__((ext_vector_type(8)))  float    v8f;

// ---------------- WMMA GEMM: C[M,N] = act(A[M,K] @ B[K,N] + bias) ----------------
// Tile: 128 (M) x 64 (N) per block of 256 threads (8 wave32s).
// Each wave: one 16-row strip x four 16x16 subtiles, K stepped by 32.
#define TM  128
#define TN  64
#define TK  32
#define LDA 40   // padded LDS stride (halves); 40*2B = 80B keeps 16B alignment
#define LDB 40
#define A_ITERS ((TM * TK) / 256)   // 16 elements per thread
#define B_ITERS ((TK * TN) / 256)   // 8 elements per thread

// Stage one K-tile into LDS. Branchless: clamped addresses + value select,
// loads fully unrolled into registers first so they issue back-to-back.
template <bool KG>
__device__ __forceinline__ void stage_tiles(const float* __restrict__ A,
                                            const float* __restrict__ B,
                                            _Float16* __restrict__ lA,
                                            _Float16* __restrict__ lB,
                                            int tid, int row0, int col0, int k0,
                                            int M, int N, int K)
{
    float av[A_ITERS];
    float bv[B_ITERS];
    #pragma unroll
    for (int it = 0; it < A_ITERS; ++it) {
        int i = tid + it * 256;
        int r = i >> 5, c = i & 31;
        int gr = row0 + r, gc = k0 + c;
        int cr = gr < M ? gr : M - 1;
        int cc = KG ? (gc < K ? gc : K - 1) : gc;
        av[it] = A[(long)cr * K + cc];
    }
    #pragma unroll
    for (int it = 0; it < B_ITERS; ++it) {
        int i = tid + it * 256;
        int k = i >> 6, n = i & 63;
        int gk = k0 + k;
        int ck = KG ? (gk < K ? gk : K - 1) : gk;
        bv[it] = B[(long)ck * N + col0 + n];
    }
    #pragma unroll
    for (int it = 0; it < A_ITERS; ++it) {
        int i = tid + it * 256;
        int r = i >> 5, c = i & 31;
        bool ok = (row0 + r < M) && (!KG || (k0 + c < K));
        lA[r * LDA + c] = (_Float16)(ok ? av[it] : 0.0f);
    }
    #pragma unroll
    for (int it = 0; it < B_ITERS; ++it) {
        int i = tid + it * 256;
        int k = i >> 6, n = i & 63;
        bool ok = !KG || (k0 + k < K);
        lB[n * LDB + k] = (_Float16)(ok ? bv[it] : 0.0f);
    }
}

__device__ __forceinline__ void wmma_step(const _Float16* __restrict__ lA,
                                          const _Float16* __restrict__ lB,
                                          int wave, int hl, int hi, v8f acc[4])
{
    // A fragment per ISA 16-bit A 16x32 layout:
    // lanes 0-15: row=hl, K 0..7 & 16..23 ; lanes 16-31: row=hl, K 8..15 & 24..31
    const _Float16* ap = &lA[(wave * 16 + hl) * LDA + hi * 8];
    v8h a_lo = *(const v8h*)(ap);
    v8h a_hi = *(const v8h*)(ap + 16);
    v16h afrag;
    #pragma unroll
    for (int i = 0; i < 8; ++i) { afrag[i] = a_lo[i]; afrag[8 + i] = a_hi[i]; }

    #pragma unroll
    for (int s = 0; s < 4; ++s) {
        const _Float16* bp = &lB[(s * 16 + hl) * LDB + hi * 8];
        v8h b_lo = *(const v8h*)(bp);
        v8h b_hi = *(const v8h*)(bp + 16);
        v16h bfrag;
        #pragma unroll
        for (int i = 0; i < 8; ++i) { bfrag[i] = b_lo[i]; bfrag[8 + i] = b_hi[i]; }
        acc[s] = __builtin_amdgcn_wmma_f32_16x16x32_f16(
            false, afrag, false, bfrag, (short)0, acc[s], false, false);
    }
}

__global__ __launch_bounds__(256)
void gemm_f16_wmma(const float* __restrict__ A, const float* __restrict__ B,
                   const float* __restrict__ bias, float* __restrict__ C,
                   int M, int N, int K, int doRelu)
{
    __shared__ _Float16 lA[TM * LDA];
    __shared__ _Float16 lB[TN * LDB];

    const int tid  = threadIdx.x;
    const int lane = tid & 31;
    const int wave = tid >> 5;
    const int row0 = blockIdx.y * TM;
    const int col0 = blockIdx.x * TN;
    const int hl   = lane & 15;   // row (A) / col (B) within 16
    const int hi   = lane >> 4;   // K-half selector per ISA layout

    v8f acc[4];
    #pragma unroll
    for (int s = 0; s < 4; ++s)
        #pragma unroll
        for (int i = 0; i < 8; ++i) acc[s][i] = 0.0f;

    const int kfull = K & ~(TK - 1);   // guard-free main loop bound

    for (int k0 = 0; k0 < kfull; k0 += TK) {
        stage_tiles<false>(A, B, lA, lB, tid, row0, col0, k0, M, N, K);
        // prefetch this thread's first A element of the next tile (global_prefetch_b8)
        {
            int r = tid >> 5, c = tid & 31;
            int gr = row0 + r;  int cr = gr < M ? gr : M - 1;
            int gc = k0 + TK + c; int cc = gc < K ? gc : K - 1;
            __builtin_prefetch(&A[(long)cr * K + cc], 0, 1);
        }
        __syncthreads();
        wmma_step(lA, lB, wave, hl, hi, acc);
        __syncthreads();
    }
    if (kfull < K) {   // remainder tile (K=773 case), zero-padded
        stage_tiles<true>(A, B, lA, lB, tid, row0, col0, kfull, M, N, K);
        __syncthreads();
        wmma_step(lA, lB, wave, hl, hi, acc);
        __syncthreads();
    }

    // C layout: VGPR v: lanes 0-15 -> (M=v, N=hl); lanes 16-31 -> (M=8+v, N=hl)
    #pragma unroll
    for (int s = 0; s < 4; ++s) {
        int col = col0 + s * 16 + hl;
        #pragma unroll
        for (int v = 0; v < 8; ++v) {
            int row = row0 + wave * 16 + hi * 8 + v;
            if (row < M) {
                float val = acc[s][v];
                if (bias) val += bias[col];
                if (doRelu) val = fmaxf(val, 0.0f);
                C[(long)row * N + col] = val;
            }
        }
    }
}

// ---------------- helpers ----------------
__device__ __forceinline__ unsigned fenc(float f) {
    unsigned u = __float_as_uint(f);
    return (u & 0x80000000u) ? ~u : (u | 0x80000000u);
}
__device__ __forceinline__ float fdec(unsigned u) {
    return __uint_as_float((u & 0x80000000u) ? (u & 0x7FFFFFFFu) : ~u);
}
__device__ __forceinline__ float lrelu(float x) { return x > 0.0f ? x : 0.2f * x; }

__global__ void fill_u32(unsigned* __restrict__ p, unsigned val, long n)
{
    long i = (long)blockIdx.x * blockDim.x + threadIdx.x;
    if (i < n) p[i] = val;
}

// a_src[n,h] = <hW[n,h,:], att_src[h,:]> ; a_dst likewise
__global__ void attn_logits(const float* __restrict__ hW,
                            const float* __restrict__ att_s,
                            const float* __restrict__ att_d,
                            float* __restrict__ a_s, float* __restrict__ a_d, int D)
{
    int idx = blockIdx.x * blockDim.x + threadIdx.x;  // n*HEADS + h
    if (idx >= NNODES * HEADS) return;
    int n = idx / HEADS, h = idx % HEADS;
    const float* p  = hW + (long)n * HEADS * D + h * D;
    const float* ws = att_s + h * D;
    const float* wd = att_d + h * D;
    float ss = 0.0f, sd = 0.0f;
    for (int d = 0; d < D; ++d) { float v = p[d]; ss += v * ws[d]; sd += v * wd[d]; }
    a_s[idx] = ss; a_d[idx] = sd;
}

__global__ void edge_max(const int* __restrict__ src, const int* __restrict__ dst,
                         const float* __restrict__ a_s, const float* __restrict__ a_d,
                         unsigned* __restrict__ menc)
{
    int e = blockIdx.x * blockDim.x + threadIdx.x;
    if (e >= ETOT) return;
    int s = (e < NEDGES) ? src[e] : (e - NEDGES);
    int d = (e < NEDGES) ? dst[e] : (e - NEDGES);
    #pragma unroll
    for (int h = 0; h < HEADS; ++h) {
        float ev = lrelu(a_s[s * HEADS + h] + a_d[d * HEADS + h]);
        atomicMax(&menc[d * HEADS + h], fenc(ev));
    }
}

__global__ void edge_expsum(const int* __restrict__ src, const int* __restrict__ dst,
                            const float* __restrict__ a_s, const float* __restrict__ a_d,
                            const unsigned* __restrict__ menc,
                            float* __restrict__ exb, float* __restrict__ z)
{
    int e = blockIdx.x * blockDim.x + threadIdx.x;
    if (e >= ETOT) return;
    int s = (e < NEDGES) ? src[e] : (e - NEDGES);
    int d = (e < NEDGES) ? dst[e] : (e - NEDGES);
    #pragma unroll
    for (int h = 0; h < HEADS; ++h) {
        float ev = lrelu(a_s[s * HEADS + h] + a_d[d * HEADS + h]);
        float ex = expf(ev - fdec(menc[d * HEADS + h]));
        exb[(long)e * HEADS + h] = ex;
        atomicAdd(&z[d * HEADS + h], ex);
    }
}

// out[dst,h,:] += hW[src,h,:] * alpha ; one thread per (edge, head, 4 dims)
__global__ void edge_aggregate(const int* __restrict__ src, const int* __restrict__ dst,
                               const float* __restrict__ hW,
                               const float* __restrict__ exb, const float* __restrict__ z,
                               float* __restrict__ out, int D)
{
    long idx = (long)blockIdx.x * blockDim.x + threadIdx.x;
    int dv = D >> 2;
    long total = (long)ETOT * HEADS * dv;
    if (idx >= total) return;
    int e  = (int)(idx / (HEADS * dv));
    int r  = (int)(idx % (HEADS * dv));
    int h  = r / dv;
    int d4 = (r % dv) * 4;
    int s = (e < NEDGES) ? src[e] : (e - NEDGES);
    int d = (e < NEDGES) ? dst[e] : (e - NEDGES);
    float alpha = exb[(long)e * HEADS + h] / (z[d * HEADS + h] + 1e-16f);
    const float* hp = hW + (long)s * HEADS * D + h * D + d4;
    float*       op = out + (long)d * HEADS * D + h * D + d4;
    float4 v = *(const float4*)hp;
    atomicAdd(op + 0, v.x * alpha);
    atomicAdd(op + 1, v.y * alpha);
    atomicAdd(op + 2, v.z * alpha);
    atomicAdd(op + 3, v.w * alpha);
}

__global__ void bias_relu(float* __restrict__ p, const float* __restrict__ bias, int F, long n)
{
    long i = (long)blockIdx.x * blockDim.x + threadIdx.x;
    if (i >= n) return;
    p[i] = fmaxf(p[i] + bias[i % F], 0.0f);
}

__global__ void pool(const float* __restrict__ h, const int* __restrict__ batch,
                     float* __restrict__ gsum, float* __restrict__ gcnt)
{
    int idx = blockIdx.x * blockDim.x + threadIdx.x;   // n*OUTF + f
    if (idx >= NNODES * OUTF) return;
    int n = idx / OUTF, f = idx % OUTF;
    int b = batch[n];
    atomicAdd(&gsum[b * OUTF + f], h[idx]);
    if (f == 0) atomicAdd(&gcnt[b], 1.0f);
}

__global__ void head(const float* __restrict__ gsum, const float* __restrict__ gcnt,
                     const float* __restrict__ Wc, const float* __restrict__ bc,
                     const float* __restrict__ Wf, const float* __restrict__ bf,
                     float* __restrict__ outp)
{
    int g = threadIdx.x;
    if (g >= BGRAPHS) return;
    float inv = 1.0f / fmaxf(gcnt[g], 1.0f);
    float l0 = bc[0], l1 = bc[1], cf = bf[0];
    for (int f = 0; f < OUTF; ++f) {
        float v = gsum[g * OUTF + f] * inv;
        l0 += v * Wc[f * 2 + 0];
        l1 += v * Wc[f * 2 + 1];
        cf += v * Wf[f];
    }
    outp[g * 2 + 0] = l0;                               // class_logits (B,2)
    outp[g * 2 + 1] = l1;
    outp[BGRAPHS * 2 + g] = 1.0f / (1.0f + expf(-cf));  // confidence (B,1)
}

// ---------------- launcher ----------------
extern "C" void kernel_launch(void* const* d_in, const int* in_sizes, int n_in,
                              void* d_out, int out_size, void* d_ws, size_t ws_size,
                              hipStream_t stream)
{
    const float* x      = (const float*)d_in[0];
    const int*   ei     = (const int*)  d_in[1];
    const int*   batch  = (const int*)  d_in[2];
    const float* W_in   = (const float*)d_in[3];
    const float* b_in   = (const float*)d_in[4];
    const float* W1     = (const float*)d_in[5];
    const float* as1    = (const float*)d_in[6];
    const float* ad1    = (const float*)d_in[7];
    const float* b1     = (const float*)d_in[8];
    const float* W2     = (const float*)d_in[9];
    const float* as2    = (const float*)d_in[10];
    const float* ad2    = (const float*)d_in[11];
    const float* b2     = (const float*)d_in[12];
    const float* W_cls  = (const float*)d_in[13];
    const float* b_cls  = (const float*)d_in[14];
    const float* W_conf = (const float*)d_in[15];
    const float* b_conf = (const float*)d_in[16];
    const int* src = ei;            // edge_index[0]
    const int* dst = ei + NEDGES;   // edge_index[1]

    // ---- workspace layout (~170.5 MB, buffers reused across layers) ----
    char* ws = (char*)d_ws;
    float* h0   = (float*)(ws + 0L);              // N*256 f32 (51.2 MB)
    float* hW1  = (float*)(ws + 51200000L);       // N*256 f32
    float* out1 = (float*)(ws + 102400000L);      // N*256 f32 -> h1 in place
    float* hW2  = h0;                             // reuse: h0 only feeds GEMM of hW1 -> dead
    float* out2 = hW1;                            // reuse: hW1 dead after layer-1 aggregation
    char*  sb   = ws + 153600000L;
    float*    a_s  = (float*)   (sb + 0L);        // N*4
    float*    a_d  = (float*)   (sb + 800000L);
    unsigned* menc = (unsigned*)(sb + 1600000L);
    float*    z    = (float*)   (sb + 2400000L);
    float*    exb  = (float*)   (sb + 3200000L);  // ETOT*4 f32 (13.6 MB)
    float*    gsum = (float*)   (sb + 16800256L); // 64*128
    float*    gcnt = (float*)   (sb + 16833024L); // 64

    const unsigned ENC_NEG_MAX = 0x00800000u;     // fenc(-FLT_MAX)
    const int T = 256;
    #define CDIV(a,b) (int)(((long)(a) + (b) - 1) / (b))

    // h0 = relu(x @ W_in + b_in)            [50000 x 773 x 256]
    gemm_f16_wmma<<<dim3(HIDF / TN, CDIV(NNODES, TM)), T, 0, stream>>>(
        x, W_in, b_in, h0, NNODES, HIDF, INDIM, 1);

    // ---------------- GAT layer 1 (H=4, D=64) ----------------
    gemm_f16_wmma<<<dim3(HIDF / TN, CDIV(NNODES, TM)), T, 0, stream>>>(
        h0, W1, nullptr, hW1, NNODES, HIDF, HIDF, 0);
    attn_logits<<<CDIV(NNODES * HEADS, T), T, 0, stream>>>(hW1, as1, ad1, a_s, a_d, DIM1);
    fill_u32<<<CDIV(NNODES * HEADS, T), T, 0, stream>>>(menc, ENC_NEG_MAX, (long)NNODES * HEADS);
    fill_u32<<<CDIV(NNODES * HEADS, T), T, 0, stream>>>((unsigned*)z, 0u, (long)NNODES * HEADS);
    fill_u32<<<CDIV(NNODES * HIDF, T), T, 0, stream>>>((unsigned*)out1, 0u, (long)NNODES * HIDF);
    edge_max<<<CDIV(ETOT, T), T, 0, stream>>>(src, dst, a_s, a_d, menc);
    edge_expsum<<<CDIV(ETOT, T), T, 0, stream>>>(src, dst, a_s, a_d, menc, exb, z);
    edge_aggregate<<<CDIV((long)ETOT * HEADS * (DIM1 / 4), T), T, 0, stream>>>(
        src, dst, hW1, exb, z, out1, DIM1);
    bias_relu<<<CDIV(NNODES * HIDF, T), T, 0, stream>>>(out1, b1, HIDF, (long)NNODES * HIDF);

    // ---------------- GAT layer 2 (H=4, D=32) ----------------
    gemm_f16_wmma<<<dim3(OUTF / TN, CDIV(NNODES, TM)), T, 0, stream>>>(
        out1, W2, nullptr, hW2, NNODES, OUTF, HIDF, 0);
    attn_logits<<<CDIV(NNODES * HEADS, T), T, 0, stream>>>(hW2, as2, ad2, a_s, a_d, DIM2);
    fill_u32<<<CDIV(NNODES * HEADS, T), T, 0, stream>>>(menc, ENC_NEG_MAX, (long)NNODES * HEADS);
    fill_u32<<<CDIV(NNODES * HEADS, T), T, 0, stream>>>((unsigned*)z, 0u, (long)NNODES * HEADS);
    fill_u32<<<CDIV(NNODES * OUTF, T), T, 0, stream>>>((unsigned*)out2, 0u, (long)NNODES * OUTF);
    edge_max<<<CDIV(ETOT, T), T, 0, stream>>>(src, dst, a_s, a_d, menc);
    edge_expsum<<<CDIV(ETOT, T), T, 0, stream>>>(src, dst, a_s, a_d, menc, exb, z);
    edge_aggregate<<<CDIV((long)ETOT * HEADS * (DIM2 / 4), T), T, 0, stream>>>(
        src, dst, hW2, exb, z, out2, DIM2);
    bias_relu<<<CDIV(NNODES * OUTF, T), T, 0, stream>>>(out2, b2, OUTF, (long)NNODES * OUTF);

    // ---------------- mean pool + heads ----------------
    fill_u32<<<CDIV(BGRAPHS * OUTF + BGRAPHS, T), T, 0, stream>>>(
        (unsigned*)gsum, 0u, (long)BGRAPHS * OUTF + BGRAPHS);
    pool<<<CDIV(NNODES * OUTF, T), T, 0, stream>>>(out2, batch, gsum, gcnt);
    head<<<1, BGRAPHS, 0, stream>>>(gsum, gcnt, W_cls, b_cls, W_conf, b_conf, (float*)d_out);

    (void)in_sizes; (void)n_in; (void)out_size; (void)ws_size;
    #undef CDIV
}
